// EtudeRNN_10806137717292
// MI455X (gfx1250) — matrix operand order repaired and checked by reading, not compiled, not run on hardware
//
#include <hip/hip_runtime.h>
#include <hip/hip_bf16.h>
#include <stddef.h>

typedef __bf16 v16bf __attribute__((ext_vector_type(16)));
typedef float  v8f   __attribute__((ext_vector_type(8)));

#define NBLK 128
#define TPB  64
#define WARM 64

// ---------------- workspace layout (bytes) ----------------
static constexpr size_t OFF_PK0   = 0;                          // L0 packed: 4096 x 1280 bf16 (rows gate-permuted)
static constexpr size_t SZ_PK0    = (size_t)4096 * 1280 * 2;
static constexpr size_t OFF_PK1   = OFF_PK0 + SZ_PK0;           // L1 packed: 4096 x 2048 bf16
static constexpr size_t SZ_PK12   = (size_t)4096 * 2048 * 2;
static constexpr size_t OFF_PK2   = OFF_PK1 + SZ_PK12;          // L2 packed
static constexpr size_t OFF_PKFC  = OFF_PK2 + SZ_PK12;          // FC packed: 256 x 1024 bf16 (identity rows)
static constexpr size_t SZ_PKFC   = (size_t)256 * 1024 * 2;
static constexpr size_t OFF_VIN0  = OFF_PKFC + SZ_PKFC;         // bf16[1280]: [x_t ; h0]
static constexpr size_t OFF_VIN1  = OFF_VIN0 + 4096;            // bf16[2048]: [h0new ; h1]
static constexpr size_t OFF_VIN2  = OFF_VIN1 + 4096;            // bf16[2048]: [h1new ; h2]
static constexpr size_t OFF_VFC   = OFF_VIN2 + 4096;            // bf16[1024]: h2new
static constexpr size_t OFF_BCOMB = OFF_VFC + 4096;             // f32[3*4096]: permuted b_ih+b_hh
static constexpr size_t OFF_C     = OFF_BCOMB + 49152;          // f32[3*1024]
static constexpr size_t OFF_BAR   = OFF_C + 12288;              // int[2]: count, gen

// ---------------- helpers ----------------
__device__ __forceinline__ unsigned short f2bf(float f) {
    unsigned int u = __float_as_uint(f);
    u += 0x7FFFu + ((u >> 16) & 1u);          // round-to-nearest-even
    return (unsigned short)(u >> 16);
}
__device__ __forceinline__ float sigm(float v) { return 1.f / (1.f + __expf(-v)); }

// device-wide sense-reversing barrier (all NBLK blocks co-resident)
__device__ __forceinline__ void grid_sync(int* cnt, int* gen, int* sgen) {
    __syncthreads();
    if (threadIdx.x == 0) {
        __threadfence();                      // publish our writes device-wide
        int g = *sgen;
        if (atomicAdd(cnt, 1) == NBLK - 1) {
            __hip_atomic_store(cnt, 0, __ATOMIC_RELAXED, __HIP_MEMORY_SCOPE_AGENT);
            __hip_atomic_store(gen, g + 1, __ATOMIC_RELEASE, __HIP_MEMORY_SCOPE_AGENT);
        } else {
            while (__hip_atomic_load(gen, __ATOMIC_ACQUIRE, __HIP_MEMORY_SCOPE_AGENT) < g + 1) {
                __builtin_amdgcn_s_sleep(2);
            }
        }
        *sgen = g + 1;
        __threadfence();                      // invalidate WGP caches before re-reading
    }
    __syncthreads();
}

// one 16-row GEMV tile over K = KT*32, B-vector broadcast across all 16 columns.
template <int KT>
__device__ __forceinline__ v8f gemv_tile(const unsigned short* __restrict__ wt,
                                         const unsigned short* __restrict__ vin,
                                         int lane) {
    const unsigned short* wl = wt + lane * 16;           // 32 B per lane per k-tile
    const unsigned short* bl = vin + ((lane >> 4) << 4); // lanes 0-15: K 0..15, 16-31: K 16..31
    v8f acc0 = {}, acc1 = {};
#pragma unroll 4
    for (int kt = 0; kt < KT; kt += 2) {
        v16bf a0 = *(const v16bf*)(wl);
        v16bf b0 = *(const v16bf*)(bl);
        v16bf a1 = *(const v16bf*)(wl + 512);
        v16bf b1 = *(const v16bf*)(bl + 32);
        wl += 1024; bl += 64;
        acc0 = __builtin_amdgcn_wmma_f32_16x16x32_bf16(false, a0, false, b0, (short)0, acc0, false, false);
        acc1 = __builtin_amdgcn_wmma_f32_16x16x32_bf16(false, a1, false, b1, (short)0, acc1, false, false);
    }
    v8f r;
#pragma unroll
    for (int i = 0; i < 8; ++i) r[i] = acc0[i] + acc1[i];
    return r;
}

// packed-row -> original-row map for gate-permuted layers:
// tile mt holds units [4*mt, 4*mt+4); within tile v = 0..15: unit = 4*mt + (v>>2), gate = v&3,
// original row r = gate*1024 + unit  (PyTorch gate order i,f,g,o).
__device__ __forceinline__ int perm_row(int mt, int v) {
    return (v & 3) * 1024 + mt * 4 + (v >> 2);
}

// ---------------- weight packing into WMMA 16-bit A layout ----------------
// dst tile (mt,kt) = 512 bf16, lane-major (32 lanes x 16 elems).
// lane<16 : v=lane,    K(j) = kt*32 +      j
// lane>=16: v=lane-16, K(j) = kt*32 + 16 + j
__global__ void pack_weights(const float* __restrict__ A, const float* __restrict__ B,
                             int Ka, int Kb, int Mtiles, int perm,
                             unsigned short* __restrict__ dst) {
    const int K  = Ka + Kb;
    const int KT = K >> 5;
    const size_t total = (size_t)Mtiles * 16 * K;
    for (size_t idx = (size_t)blockIdx.x * blockDim.x + threadIdx.x; idx < total;
         idx += (size_t)gridDim.x * blockDim.x) {
        int tile = (int)(idx >> 9);
        int e    = (int)(idx & 511);
        int lane = e >> 4;
        int j    = e & 15;
        int mt   = tile / KT;
        int kt   = tile - mt * KT;
        int v    = lane & 15;
        int row  = perm ? perm_row(mt, v) : (mt * 16 + v);
        int k    = kt * 32 + ((lane >> 4) << 4) + j;
        float w  = (k < Ka) ? A[(size_t)row * Ka + k] : B[(size_t)row * Kb + (k - Ka)];
        dst[idx] = f2bf(w);
    }
}

// ---------------- per-call state init ----------------
__global__ void init_state(const float* __restrict__ x, const float* __restrict__ h0,
                           const float* __restrict__ c0, const float* __restrict__ b_ih,
                           const float* __restrict__ b_hh, const int* __restrict__ tsteps,
                           float* __restrict__ out, char* __restrict__ ws) {
    unsigned short* vin0  = (unsigned short*)(ws + OFF_VIN0);
    unsigned short* vin1  = (unsigned short*)(ws + OFF_VIN1);
    unsigned short* vin2  = (unsigned short*)(ws + OFF_VIN2);
    float*          bcomb = (float*)(ws + OFF_BCOMB);
    float*          cbuf  = (float*)(ws + OFF_C);
    int*            bar   = (int*)(ws + OFF_BAR);
    const int T = tsteps[0];
    const int tid = blockIdx.x * blockDim.x + threadIdx.x;
    const int stride = gridDim.x * blockDim.x;
    if (tid == 0) { bar[0] = 0; bar[1] = 0; }
    for (int i = tid; i < 3072; i += stride) cbuf[i] = c0[i];
    // combined, gate-permuted biases
    for (int p = tid; p < 3 * 4096; p += stride) {
        int l  = p >> 12;
        int pp = p & 4095;
        int r  = perm_row(pp >> 4, pp & 15);
        bcomb[p] = b_ih[l * 4096 + r] + b_hh[l * 4096 + r];
    }
    for (int i = tid; i < 1024; i += stride) {
        vin0[256 + i]  = f2bf(h0[i]);          // layer0 recurrent h
        vin1[1024 + i] = f2bf(h0[1024 + i]);   // layer1 recurrent h
        vin2[1024 + i] = f2bf(h0[2048 + i]);   // layer2 recurrent h
    }
    for (int d = tid; d < 256; d += stride) {
        float v = x[d * WARM + 0];
        vin0[d] = f2bf(v);                     // first input
        out[(size_t)d * T + 0] = v;            // out[:,0] = x[:,0]
    }
}

// ---------------- persistent LSTM kernel: all T steps in one launch ----------------
__global__ void __launch_bounds__(TPB)
lstm_persistent(const float* __restrict__ x, const float* __restrict__ b_fc,
                const int* __restrict__ tsteps, float* __restrict__ out,
                char* __restrict__ ws) {
    const unsigned short* pk0  = (const unsigned short*)(ws + OFF_PK0);
    const unsigned short* pk1  = (const unsigned short*)(ws + OFF_PK1);
    const unsigned short* pk2  = (const unsigned short*)(ws + OFF_PK2);
    const unsigned short* pkfc = (const unsigned short*)(ws + OFF_PKFC);
    unsigned short* vin0  = (unsigned short*)(ws + OFF_VIN0);
    unsigned short* vin1  = (unsigned short*)(ws + OFF_VIN1);
    unsigned short* vin2  = (unsigned short*)(ws + OFF_VIN2);
    unsigned short* vfc   = (unsigned short*)(ws + OFF_VFC);
    const float*    bcomb = (const float*)(ws + OFF_BCOMB);
    float*          cbuf  = (float*)(ws + OFF_C);
    int*            bar   = (int*)(ws + OFF_BAR);

    __shared__ int sgen;
    if (threadIdx.x == 0) sgen = 0;
    __syncthreads();

    const int lane = threadIdx.x & 31;
    const int wid  = threadIdx.x >> 5;
    const int gw   = blockIdx.x * (TPB / 32) + wid;          // 0..255 == M-tile id
    const int T    = tsteps[0];

    for (int t = 0; t + 1 < T; ++t) {
#pragma unroll
        for (int l = 0; l < 3; ++l) {
            // ---- gate GEMV: one 16-row (4-unit) tile per wave, full K per wave ----
            v8f acc;
            if (l == 0)      acc = gemv_tile<40>(pk0 + (size_t)gw * 40 * 512, vin0, lane);
            else if (l == 1) acc = gemv_tile<64>(pk1 + (size_t)gw * 64 * 512, vin1, lane);
            else             acc = gemv_tile<64>(pk2 + (size_t)gw * 64 * 512, vin2, lane);
            // lane 0 holds packed rows v=0..7 (units 4g+0,4g+1), lane 16 v=8..15 (units 4g+2,4g+3);
            // each packed row quad = gates {i,f,g,o} of one unit -> fused local cell update.
            if ((lane & 15) == 0) {
                const int pbase = gw * 16 + (lane >> 4) * 8;   // packed row base for this lane
#pragma unroll
                for (int q = 0; q < 2; ++q) {
                    const int u = gw * 4 + (lane >> 4) * 2 + q;
                    const float ig = acc[q * 4 + 0] + bcomb[l * 4096 + pbase + q * 4 + 0];
                    const float fg = acc[q * 4 + 1] + bcomb[l * 4096 + pbase + q * 4 + 1];
                    const float gg = acc[q * 4 + 2] + bcomb[l * 4096 + pbase + q * 4 + 2];
                    const float og = acc[q * 4 + 3] + bcomb[l * 4096 + pbase + q * 4 + 3];
                    const float cn = sigm(fg) * cbuf[l * 1024 + u] + sigm(ig) * tanhf(gg);
                    const float hn = sigm(og) * tanhf(cn);
                    cbuf[l * 1024 + u] = cn;
                    const unsigned short hb = f2bf(hn);
                    if (l == 0)      { vin1[u] = hb; vin0[256 + u]  = hb; }
                    else if (l == 1) { vin2[u] = hb; vin1[1024 + u] = hb; }
                    else             { vfc[u]  = hb; vin2[1024 + u] = hb; }
                }
            }
            grid_sync(&bar[0], &bar[1], &sgen);
        }

        // ---- FC + sigmoid + teacher-forcing select + output (waves 0..15) ----
        if (gw < 16) {
            v8f acc = gemv_tile<32>(pkfc + (size_t)gw * 32 * 512, vfc, lane);
            if ((lane & 15) == 0) {
                const int rbase = gw * 16 + (lane >> 4) * 8;
#pragma unroll
                for (int v = 0; v < 8; ++v) {
                    const int r = rbase + v;
                    float nxt;
                    if (t + 1 < WARM) nxt = x[r * WARM + (t + 1)];
                    else              nxt = sigm(acc[v] + b_fc[r]);
                    out[(size_t)r * T + (t + 1)] = nxt;
                    vin0[r] = f2bf(nxt);
                }
            }
        }
        grid_sync(&bar[0], &bar[1], &sgen);
    }
}

// ---------------- host entry ----------------
extern "C" void kernel_launch(void* const* d_in, const int* in_sizes, int n_in,
                              void* d_out, int out_size, void* d_ws, size_t ws_size,
                              hipStream_t stream) {
    const float* x     = (const float*)d_in[0];
    const float* h0    = (const float*)d_in[1];
    const float* c0    = (const float*)d_in[2];
    const float* W_ih0 = (const float*)d_in[3];
    const float* W_hh0 = (const float*)d_in[4];
    const float* W_ih1 = (const float*)d_in[5];
    const float* W_hh1 = (const float*)d_in[6];
    const float* W_ih2 = (const float*)d_in[7];
    const float* W_hh2 = (const float*)d_in[8];
    const float* b_ih  = (const float*)d_in[9];
    const float* b_hh  = (const float*)d_in[10];
    const float* W_fc  = (const float*)d_in[11];
    const float* b_fc  = (const float*)d_in[12];
    const int*   ts    = (const int*)d_in[13];
    float* out = (float*)d_out;
    char*  ws  = (char*)d_ws;

    unsigned short* pk0  = (unsigned short*)(ws + OFF_PK0);
    unsigned short* pk1  = (unsigned short*)(ws + OFF_PK1);
    unsigned short* pk2  = (unsigned short*)(ws + OFF_PK2);
    unsigned short* pkfc = (unsigned short*)(ws + OFF_PKFC);

    // one-time (per call) bf16 re-pack of all weights into WMMA A-tile layout,
    // with gate rows permuted so each tile owns 4 complete hidden units.
    pack_weights<<<4096, 256, 0, stream>>>(W_ih0, W_hh0, 256, 1024, 256, 1, pk0);
    pack_weights<<<4096, 256, 0, stream>>>(W_ih1, W_hh1, 1024, 1024, 256, 1, pk1);
    pack_weights<<<4096, 256, 0, stream>>>(W_ih2, W_hh2, 1024, 1024, 256, 1, pk2);
    pack_weights<<<1024, 256, 0, stream>>>(W_fc, nullptr, 1024, 0, 16, 0, pkfc);

    init_state<<<8, 512, 0, stream>>>(x, h0, c0, b_ih, b_hh, ts, out, ws);

    lstm_persistent<<<NBLK, TPB, 0, stream>>>(x, b_fc, ts, out, ws);
}
